// lstm_seq2seq_73005854098247
// MI455X (gfx1250) — compile-verified, matrix operand order checked by reading, not run on hardware
//
#include <hip/hip_runtime.h>
#include <hip/hip_bf16.h>

typedef __attribute__((ext_vector_type(16))) _Float16 v16h;
typedef __attribute__((ext_vector_type(8)))  _Float16 v8h;
typedef __attribute__((ext_vector_type(8)))  float    v8f;

#define SEQ   20
#define BATCH 16384
#define NINP  2
#define HID   256
#define EMB   64
#define TLEN  30

// ---------------------------------------------------------------------------
// helpers
// ---------------------------------------------------------------------------
__device__ __forceinline__ float sigmoidf_(float x) {
  return 1.0f / (1.0f + __expf(-x));
}

// Load a 16x32 f16 WMMA A/B fragment for one lane.
// Row-major [rows, ldK] source; lane holds row (lane&15); hi = lane>>4.
// Per ISA layout: element j -> K = (j<8) ? hi*8+j : 16+hi*8+(j-8).
// p must point at rowptr + kb*32 + hi*8 (16B aligned since ldK % 8 == 0).
__device__ __forceinline__ v16h load_frag16(const _Float16* p) {
  v8h lo = *(const v8h*)p;
  v8h hh = *(const v8h*)(p + 16);
  v16h r;
#pragma unroll
  for (int j = 0; j < 8; ++j) { r[j] = lo[j]; r[j + 8] = hh[j]; }
  return r;
}

// ---------------------------------------------------------------------------
// small utility kernels
// ---------------------------------------------------------------------------
__global__ void zero_f32_kernel(float* __restrict__ p, int n) {
  int i = blockIdx.x * blockDim.x + threadIdx.x;
  if (i < n) p[i] = 0.0f;
}

__global__ void zero_f16_kernel(_Float16* __restrict__ p, int n) {
  int i = blockIdx.x * blockDim.x + threadIdx.x;
  if (i < n) p[i] = (_Float16)0.0f;
}

__global__ void cvt_f16_kernel(const float* __restrict__ in,
                               _Float16* __restrict__ out, int n) {
  int i = blockIdx.x * blockDim.x + threadIdx.x;
  if (i < n) out[i] = (_Float16)in[i];
}

// x_emb[s,b,e] = x[s,b,0]*W_emb[e,0] + x[s,b,1]*W_emb[e,1] + b_emb[e]  (K=2)
__global__ void emb_kernel(const float* __restrict__ x,      // [SEQ,B,2]
                           const float* __restrict__ Wemb,   // [64,2]
                           const float* __restrict__ bemb,   // [64]
                           _Float16* __restrict__ xemb) {    // [SEQ,B,64] f16
  int id = blockIdx.x * blockDim.x + threadIdx.x;  // < SEQ*B*EMB
  int e  = id & (EMB - 1);
  int sb = id >> 6;
  float x0 = x[(size_t)sb * 2 + 0];
  float x1 = x[(size_t)sb * 2 + 1];
  float v  = x0 * Wemb[e * 2 + 0] + x1 * Wemb[e * 2 + 1] + bemb[e];
  xemb[id] = (_Float16)v;
}

// ---------------------------------------------------------------------------
// fused epilogue for one 16x16 C tile: bias (+ INP=2 scalar path),
// sigmoid/tanh gate math, c/h update, h written as f32 and f16.
// ---------------------------------------------------------------------------
__device__ __forceinline__ void lstm_epilogue_tile(
    v8f gi_, v8f gf_, v8f gg_, v8f go_,
    int row0, int col, int hi,
    float bi, float bf, float bg, float bo,
    const float* __restrict__ dx,
    float wi0, float wi1, float wf0, float wf1,
    float wg0, float wg1, float wo0, float wo1,
    float* __restrict__ c, float* __restrict__ h32,
    _Float16* __restrict__ hout) {
#pragma unroll
  for (int r = 0; r < 8; ++r) {
    // C/D layout: VGPR r holds M = hi*8 + r, N = lane&15
    const int row = row0 + hi * 8 + r;
    float gi = gi_[r] + bi;
    float gf = gf_[r] + bf;
    float gg = gg_[r] + bg;
    float go = go_[r] + bo;
    if (dx) {
      const float x0 = dx[(size_t)row * 2 + 0];
      const float x1 = dx[(size_t)row * 2 + 1];
      gi += x0 * wi0 + x1 * wi1;
      gf += x0 * wf0 + x1 * wf1;
      gg += x0 * wg0 + x1 * wg1;
      go += x0 * wo0 + x1 * wo1;
    }
    const size_t idx = (size_t)row * HID + col;
    const float cn = sigmoidf_(gf) * c[idx] + sigmoidf_(gi) * tanhf(gg);
    const float hn = sigmoidf_(go) * tanhf(cn);
    c[idx]    = cn;
    h32[idx]  = hn;
    hout[idx] = (_Float16)hn;
  }
}

// ---------------------------------------------------------------------------
// fused LSTM step: gates = xin@Wih^T + hin@Whh^T + bias (+ dec scalar x-path),
// then c/h update in the epilogue. One block = 128 batch rows x 16 hidden
// cols, 4 waves; each wave computes a 32x16 tile: two A fragments share the
// four B (gate) fragments -> 8 WMMA per 12 b128 loads per K-chunk.
// ---------------------------------------------------------------------------
__global__ __launch_bounds__(128)
void lstm_step_kernel(const _Float16* __restrict__ xin,     // [B,xK] f16 or null
                      int xK,
                      const _Float16* __restrict__ Wih,     // [4H,xK] f16 or null
                      const _Float16* __restrict__ Whh,     // [4H,HID] f16
                      const _Float16* __restrict__ hin,     // [B,HID] f16
                      const float* __restrict__ bias,       // [4H]
                      const float* __restrict__ dx,         // [B,2] f32 or null
                      const float* __restrict__ Wih_dec,    // [4H,2] f32 or null
                      float* __restrict__ c,                // [B,HID] in/out
                      float* __restrict__ h32,              // [B,HID] out
                      _Float16* __restrict__ hout) {        // [B,HID] out
  const int lane = threadIdx.x & 31;
  const int wave = threadIdx.x >> 5;
  const int m    = lane & 15;
  const int hi   = lane >> 4;
  const int col_base = blockIdx.x * 16;
  const int row_base = blockIdx.y * 128 + wave * 32;  // 2 row tiles per wave

  v8f i0 = {}, i1 = {}, f0 = {}, f1 = {};
  v8f g0 = {}, g1 = {}, o0 = {}, o1 = {};

  // ---- input-projection GEMM (encoder path, xK = EMB = 64) ----
  if (xin) {
    const int nk = xK >> 5;
    for (int kb = 0; kb < nk; ++kb) {
      const int ko = kb * 32 + hi * 8;
      v16h a0 = load_frag16(xin + (size_t)(row_base + m) * xK + ko);
      v16h a1 = load_frag16(xin + (size_t)(row_base + 16 + m) * xK + ko);
      const size_t wro = (size_t)(col_base + m) * xK + ko;
      v16h b0 = load_frag16(Wih + wro);
      v16h b1 = load_frag16(Wih + (size_t)1 * HID * xK + wro);
      v16h b2 = load_frag16(Wih + (size_t)2 * HID * xK + wro);
      v16h b3 = load_frag16(Wih + (size_t)3 * HID * xK + wro);
      i0 = __builtin_amdgcn_wmma_f32_16x16x32_f16(false, a0, false, b0, (short)0, i0, false, false);
      i1 = __builtin_amdgcn_wmma_f32_16x16x32_f16(false, a1, false, b0, (short)0, i1, false, false);
      f0 = __builtin_amdgcn_wmma_f32_16x16x32_f16(false, a0, false, b1, (short)0, f0, false, false);
      f1 = __builtin_amdgcn_wmma_f32_16x16x32_f16(false, a1, false, b1, (short)0, f1, false, false);
      g0 = __builtin_amdgcn_wmma_f32_16x16x32_f16(false, a0, false, b2, (short)0, g0, false, false);
      g1 = __builtin_amdgcn_wmma_f32_16x16x32_f16(false, a1, false, b2, (short)0, g1, false, false);
      o0 = __builtin_amdgcn_wmma_f32_16x16x32_f16(false, a0, false, b3, (short)0, o0, false, false);
      o1 = __builtin_amdgcn_wmma_f32_16x16x32_f16(false, a1, false, b3, (short)0, o1, false, false);
    }
  }

  // ---- recurrent GEMM over HID = 256 (8 K-chunks) ----
#pragma unroll
  for (int kb = 0; kb < HID / 32; ++kb) {
    const int ko = kb * 32 + hi * 8;
    v16h a0 = load_frag16(hin + (size_t)(row_base + m) * HID + ko);
    v16h a1 = load_frag16(hin + (size_t)(row_base + 16 + m) * HID + ko);
    const size_t wro = (size_t)(col_base + m) * HID + ko;
    v16h b0 = load_frag16(Whh + wro);
    v16h b1 = load_frag16(Whh + (size_t)1 * HID * HID + wro);
    v16h b2 = load_frag16(Whh + (size_t)2 * HID * HID + wro);
    v16h b3 = load_frag16(Whh + (size_t)3 * HID * HID + wro);
    i0 = __builtin_amdgcn_wmma_f32_16x16x32_f16(false, a0, false, b0, (short)0, i0, false, false);
    i1 = __builtin_amdgcn_wmma_f32_16x16x32_f16(false, a1, false, b0, (short)0, i1, false, false);
    f0 = __builtin_amdgcn_wmma_f32_16x16x32_f16(false, a0, false, b1, (short)0, f0, false, false);
    f1 = __builtin_amdgcn_wmma_f32_16x16x32_f16(false, a1, false, b1, (short)0, f1, false, false);
    g0 = __builtin_amdgcn_wmma_f32_16x16x32_f16(false, a0, false, b2, (short)0, g0, false, false);
    g1 = __builtin_amdgcn_wmma_f32_16x16x32_f16(false, a1, false, b2, (short)0, g1, false, false);
    o0 = __builtin_amdgcn_wmma_f32_16x16x32_f16(false, a0, false, b3, (short)0, o0, false, false);
    o1 = __builtin_amdgcn_wmma_f32_16x16x32_f16(false, a1, false, b3, (short)0, o1, false, false);
  }

  // ---- fused epilogue ----
  const int col = col_base + m;
  const float bi = bias[0 * HID + col];
  const float bf = bias[1 * HID + col];
  const float bg = bias[2 * HID + col];
  const float bo = bias[3 * HID + col];

  float wi0 = 0.f, wi1 = 0.f, wf0 = 0.f, wf1 = 0.f;
  float wg0 = 0.f, wg1 = 0.f, wo0 = 0.f, wo1 = 0.f;
  if (dx) {
    wi0 = Wih_dec[(0 * HID + col) * 2 + 0]; wi1 = Wih_dec[(0 * HID + col) * 2 + 1];
    wf0 = Wih_dec[(1 * HID + col) * 2 + 0]; wf1 = Wih_dec[(1 * HID + col) * 2 + 1];
    wg0 = Wih_dec[(2 * HID + col) * 2 + 0]; wg1 = Wih_dec[(2 * HID + col) * 2 + 1];
    wo0 = Wih_dec[(3 * HID + col) * 2 + 0]; wo1 = Wih_dec[(3 * HID + col) * 2 + 1];
  }

  lstm_epilogue_tile(i0, f0, g0, o0, row_base, col, hi, bi, bf, bg, bo,
                     dx, wi0, wi1, wf0, wf1, wg0, wg1, wo0, wo1,
                     c, h32, hout);
  lstm_epilogue_tile(i1, f1, g1, o1, row_base + 16, col, hi, bi, bf, bg, bo,
                     dx, wi0, wi1, wf0, wf1, wg0, wg1, wo0, wo1,
                     c, h32, hout);
}

// ---------------------------------------------------------------------------
// decoder output projection: out[row,:2] = h[row,:] @ W_out^T + b_out
// one wave per row, lane-parallel over HID with xor-shuffle reduction
// ---------------------------------------------------------------------------
__global__ __launch_bounds__(256)
void dec_out_kernel(const float* __restrict__ h32,    // [B,HID]
                    const float* __restrict__ Wout,   // [2,HID]
                    const float* __restrict__ bout,   // [2]
                    float* __restrict__ out) {        // [B,2]
  const int lane = threadIdx.x & 31;
  const int row  = blockIdx.x * 8 + (threadIdx.x >> 5);
  const float* hr = h32 + (size_t)row * HID;
  float s0 = 0.f, s1 = 0.f;
#pragma unroll
  for (int i = 0; i < HID / 32; ++i) {
    const int k = lane + i * 32;
    const float hv = hr[k];
    s0 += hv * Wout[k];
    s1 += hv * Wout[HID + k];
  }
#pragma unroll
  for (int off = 16; off > 0; off >>= 1) {
    s0 += __shfl_xor(s0, off, 32);
    s1 += __shfl_xor(s1, off, 32);
  }
  if (lane == 0) {
    out[(size_t)row * 2 + 0] = s0 + bout[0];
    out[(size_t)row * 2 + 1] = s1 + bout[1];
  }
}

// ---------------------------------------------------------------------------
// launch
// ---------------------------------------------------------------------------
extern "C" void kernel_launch(void* const* d_in, const int* in_sizes, int n_in,
                              void* d_out, int out_size, void* d_ws, size_t ws_size,
                              hipStream_t stream) {
  const float* x_input  = (const float*)d_in[0];   // [20,16384,2]
  const float* W_emb    = (const float*)d_in[1];   // [64,2]
  const float* b_emb    = (const float*)d_in[2];   // [64]
  const float* W_ih_enc = (const float*)d_in[3];   // [1024,64]
  const float* W_hh_enc = (const float*)d_in[4];   // [1024,256]
  const float* b_enc    = (const float*)d_in[5];   // [1024]
  const float* W_ih_dec = (const float*)d_in[6];   // [1024,2]
  const float* W_hh_dec = (const float*)d_in[7];   // [1024,256]
  const float* b_dec    = (const float*)d_in[8];   // [1024]
  const float* W_out    = (const float*)d_in[9];   // [2,256]
  const float* b_out    = (const float*)d_in[10];  // [2]
  float* out = (float*)d_out;                      // [30,16384,2]

  // workspace carve-up (256B aligned)
  char* ws = (char*)d_ws;
  size_t off = 0;
  auto carve = [&](size_t bytes) {
    char* p = ws + off;
    off = (off + bytes + 255) & ~(size_t)255;
    return p;
  };
  _Float16* xemb    = (_Float16*)carve((size_t)SEQ * BATCH * EMB * 2);  // 40 MB
  _Float16* Wih_e16 = (_Float16*)carve((size_t)4 * HID * EMB * 2);
  _Float16* Whh_e16 = (_Float16*)carve((size_t)4 * HID * HID * 2);
  _Float16* Whh_d16 = (_Float16*)carve((size_t)4 * HID * HID * 2);
  float*    c_buf   = (float*)carve((size_t)BATCH * HID * 4);           // 16 MB
  float*    h32     = (float*)carve((size_t)BATCH * HID * 4);           // 16 MB
  _Float16* h16a    = (_Float16*)carve((size_t)BATCH * HID * 2);        // 8 MB
  _Float16* h16b    = (_Float16*)carve((size_t)BATCH * HID * 2);        // 8 MB

  // ---- init + weight conversion ----
  {
    int n = BATCH * HID;
    zero_f32_kernel<<<(n + 255) / 256, 256, 0, stream>>>(c_buf, n);
    zero_f16_kernel<<<(n + 255) / 256, 256, 0, stream>>>(h16a, n);
    int n1 = 4 * HID * EMB;
    cvt_f16_kernel<<<(n1 + 255) / 256, 256, 0, stream>>>(W_ih_enc, Wih_e16, n1);
    int n2 = 4 * HID * HID;
    cvt_f16_kernel<<<(n2 + 255) / 256, 256, 0, stream>>>(W_hh_enc, Whh_e16, n2);
    cvt_f16_kernel<<<(n2 + 255) / 256, 256, 0, stream>>>(W_hh_dec, Whh_d16, n2);
    int ne = SEQ * BATCH * EMB;
    emb_kernel<<<(ne + 255) / 256, 256, 0, stream>>>(x_input, W_emb, b_emb, xemb);
  }

  const dim3 grid(HID / 16, BATCH / 128);  // (16, 128)
  const dim3 block(128);

  _Float16* hin  = h16a;
  _Float16* hout = h16b;

  // ---- encoder: 20 steps ----
  for (int t = 0; t < SEQ; ++t) {
    lstm_step_kernel<<<grid, block, 0, stream>>>(
        xemb + (size_t)t * BATCH * EMB, EMB, Wih_e16, Whh_e16,
        hin, b_enc, nullptr, nullptr, c_buf, h32, hout);
    _Float16* tmp = hin; hin = hout; hout = tmp;
  }

  // ---- autoregressive decoder: 30 steps ----
  for (int t = 0; t < TLEN; ++t) {
    const float* dx = (t == 0)
        ? x_input + (size_t)(SEQ - 1) * BATCH * NINP
        : out + (size_t)(t - 1) * BATCH * NINP;
    lstm_step_kernel<<<grid, block, 0, stream>>>(
        nullptr, 0, nullptr, Whh_d16,
        hin, b_dec, dx, W_ih_dec, c_buf, h32, hout);
    dec_out_kernel<<<BATCH / 8, 256, 0, stream>>>(
        h32, W_out, b_out, out + (size_t)t * BATCH * NINP);
    _Float16* tmp = hin; hin = hout; hout = tmp;
  }

  (void)in_sizes; (void)n_in; (void)out_size; (void)ws_size;
}
